// SphericalCNNCls_65309272703435
// MI455X (gfx1250) — matrix operand order repaired and checked by reading, not compiled, not run on hardware
//
#include <hip/hip_runtime.h>
#include <hip/hip_bf16.h>

typedef __bf16 bf16;
typedef __attribute__((ext_vector_type(16))) __bf16 v16bf;
typedef __attribute__((ext_vector_type(8)))  float  v8f;
typedef __attribute__((ext_vector_type(2)))  float  v2f;

#define ALPHA 0.3f
#define BN_EPS 1e-3f
#define SIGMA2 0.08f   // 2 * 0.2^2

// ---------------------------------------------------------------------------
// Phase 1a: Gaussian shell density via f32 WMMA (16x16x4).
//   d2[m][s] = |x_m|^2 + |c_s|^2 - 2 x_m . c_s   as one WMMA:
//   A row m = [-2px, -2py, -2pz, |p|^2], B col s = [cx, cy, cz, 1], C = |c_s|^2
// One wave per (s-tile of 16, batch b, shell); 128 WMMA+exp iterations over N.
// Branch-free inner loop; exp(-d2/sigma2) folded into a single v_exp_f32
// (exp2) with one pre-multiply.
// ---------------------------------------------------------------------------
__global__ void __launch_bounds__(32)
shell_density_kernel(const float* __restrict__ x,        // [64][2048][3]
                     const float* __restrict__ dirs,     // [3][512][3]
                     float* __restrict__ f)              // [3][64][512]
{
    const int st   = blockIdx.x;           // 0..31  (s tile)
    const int b    = blockIdx.y;           // 0..63
    const int shel = blockIdx.z;           // 0..2
    const int lane = threadIdx.x;          // 0..31
    const int half = lane >> 4;
    const int lm   = lane & 15;

    const float r = (shel == 0) ? 0.4f : ((shel == 1) ? 0.8f : 1.2f);
    const int s = st * 16 + lm;
    const float* dp = dirs + ((size_t)shel * 512 + s) * 3;
    const float cx = r * dp[0], cy = r * dp[1], cz = r * dp[2];
    const float cn2 = cx * cx + cy * cy + cz * cz;

    v2f bfrag;
    bfrag[0] = half ? cz   : cx;
    bfrag[1] = half ? 1.0f : cy;

    v8f cinit;
#pragma unroll
    for (int i = 0; i < 8; ++i) cinit[i] = cn2;

    v8f acc;
#pragma unroll
    for (int i = 0; i < 8; ++i) acc[i] = 0.0f;

    // exp(-d2/SIGMA2) == exp2(d2 * -1/(SIGMA2*ln2)) -> single mul + v_exp_f32
    const float exp2_scale = -1.4426950408889634f / SIGMA2;

    for (int n0 = 0; n0 < 2048; n0 += 16) {
        const float* xp = x + ((size_t)b * 2048 + n0 + lm) * 3;
        const float px = xp[0], py = xp[1], pz = xp[2];
        const float pn2 = px * px + py * py + pz * pz;
        v2f afrag;                                      // branch-free per-lane select
        afrag[0] = half ? (-2.0f * pz) : (-2.0f * px);
        afrag[1] = half ? pn2          : (-2.0f * py);

        v8f d2 = __builtin_amdgcn_wmma_f32_16x16x4_f32(
            false, afrag, false, bfrag, (short)0, cinit, false, false);
#pragma unroll
        for (int v = 0; v < 8; ++v)
            acc[v] += __builtin_amdgcn_exp2f(d2[v] * exp2_scale);
    }

    float s8 = 0.0f;
#pragma unroll
    for (int v = 0; v < 8; ++v) s8 += acc[v];
    s8 += __shfl_xor(s8, 16);                    // combine lane halves (M 0-7 + 8-15)
    if (half == 0)
        f[((size_t)shel * 64 + b) * 512 + s] = s8 * (1.0f / 2048.0f);
}

// ---------------------------------------------------------------------------
// Phase 1b: project shell densities onto SH basis: sh[b][j][shell] = f . A_sh[j]
// ---------------------------------------------------------------------------
__global__ void sh_project_kernel(const float* __restrict__ f,     // [3][64][512]
                                  const float* __restrict__ A_sh,  // [36][512]
                                  float* __restrict__ sh)          // [64][36][3]
{
    const int b = blockIdx.x, shel = blockIdx.y, j = threadIdx.x;
    if (j >= 36) return;
    const float* fp = f + ((size_t)shel * 64 + b) * 512;
    const float* ap = A_sh + (size_t)j * 512;
    float a = 0.0f;
    for (int k = 0; k < 512; ++k) a += fp[k] * ap[k];
    sh[((size_t)b * 36 + j) * 3 + shel] = a;
}

// ---------------------------------------------------------------------------
// Pullback index: flat Wigner index of (l, m, n=0) for SH coeff j.
// ---------------------------------------------------------------------------
__device__ __forceinline__ int pullback_t(int j) {
    int off = 0;
    for (int l = 0; l <= 5; ++l) {
        int d = 2 * l + 1;
        if (j < d) return off + j * d + l;
        j -= d; off += d * d;
    }
    return -1;
}

// Phase 1c: pullback scatter + channel mix W0 (3 -> 16), output transposed
// bf16 [64][16][288] (t >= 286 zero-padded for the K-dim of the next GEMM).
__global__ void mix0_kernel(const float* __restrict__ sh,   // [64][36][3]
                            const float* __restrict__ W0,   // [3][16]
                            const float* __restrict__ b0,   // [16]
                            bf16* __restrict__ out)         // [64][16][288]
{
    __shared__ int sidx[36];
    const int b = blockIdx.x;
    const int t = threadIdx.x;                  // 0..287
    if (t < 36) sidx[t] = pullback_t(t);
    __syncthreads();
    int j = -1;
    for (int k = 0; k < 36; ++k) if (sidx[k] == t) j = k;
    const bool valid = (t < 286);
    for (int co = 0; co < 16; ++co) {
        float a = valid ? b0[co] : 0.0f;
        if (j >= 0) {
#pragma unroll
            for (int ci = 0; ci < 3; ++ci)
                a += sh[((size_t)b * 36 + j) * 3 + ci] * W0[ci * 16 + co];
        }
        out[((size_t)b * 16 + co) * 288 + t] = (bf16)a;
    }
}

// ---------------------------------------------------------------------------
// Generic f32 -> zero-padded bf16 cast for the shared Wigner bases.
// ---------------------------------------------------------------------------
__global__ void cast_pad_kernel(const float* __restrict__ in, bf16* __restrict__ out,
                                int M, int K, int Mp, int Kp)
{
    const int i = blockIdx.x * 256 + threadIdx.x;
    if (i >= Mp * Kp) return;
    const int mp = i / Kp, kp = i % Kp;
    const float v = (mp < M && kp < K) ? in[(size_t)mp * K + kp] : 0.0f;
    out[i] = (bf16)v;
}

// ---------------------------------------------------------------------------
// Batched GEMM on v_wmma_f32_16x16x32_bf16:
//   Out[b][c][m] = sum_k A[m][k] * Bt[b][c][k]       (A shared across batch)
// A: bf16 [Mp][Kp] row-major.  Bt: bf16 [Bb][C][Kp] (activation, transposed).
// Two waves per block (m-tiles m0, m0+16) share one 16-channel B panel.
// The panel (contiguous 16*Kp bf16, <= 32KB) is staged into LDS once with
// gfx1250 async global->LDS b128 copies (ASYNCcnt + barrier), then the inner
// loop reads B fragments via ds_load_b128 while A streams from L2.
// Two independent accumulators break the WMMA D->C dependency chain.
// grid = (Mp/32, C/16, Bb), block = 64.
// ---------------------------------------------------------------------------
__global__ void __launch_bounds__(64)
gemm_bf16_kernel(const bf16* __restrict__ A,
                 const bf16* __restrict__ Bt,
                 float* __restrict__ Out,
                 int Mp, int Kp, int C)
{
    __shared__ bf16 sB[16 * 1024];               // max panel: 16 x 1024 bf16 = 32KB

    const int wave = threadIdx.x >> 5;
    const int lane = threadIdx.x & 31;
    const int m0   = (blockIdx.x * 2 + wave) * 16;
    const int c0   = blockIdx.y * 16;
    const int b    = blockIdx.z;
    const int half = lane >> 4;
    const int lm   = lane & 15;

    // ---- async-stage the contiguous B panel (rows c0..c0+15) into LDS ----
    const bf16* panel = Bt + ((size_t)b * C + c0) * Kp;
    const int nchunks = (16 * Kp * 2) / 16;      // 16-byte chunks = 2*Kp
    const unsigned lds_base = (unsigned)(uintptr_t)(&sB[0]);   // low 32 bits = LDS offset
    for (int ci = (int)threadIdx.x; ci < nchunks; ci += 64) {
        const unsigned ldsa = lds_base + (unsigned)ci * 16u;
        const unsigned goff = (unsigned)ci * 16u;
        asm volatile("global_load_async_to_lds_b128 %0, %1, %2"
                     :: "v"(ldsa), "v"(goff), "s"(panel) : "memory");
    }
    asm volatile("s_wait_asynccnt 0" ::: "memory");
    __syncthreads();

    const bf16* Arow = A  + (size_t)(m0 + lm) * Kp + 8 * half;   // global
    const bf16* Brow = sB + (size_t)lm * Kp + 8 * half;          // LDS

    v8f acc0, acc1;
#pragma unroll
    for (int i = 0; i < 8; ++i) { acc0[i] = 0.0f; acc1[i] = 0.0f; }

    union Frag { uint4 u[2]; v16bf v; };
    int k = 0;
    for (; k + 64 <= Kp; k += 64) {              // two independent WMMA chains
        Frag ua0, ub0, ua1, ub1;
        ua0.u[0] = *(const uint4*)(Arow + k);
        ua0.u[1] = *(const uint4*)(Arow + k + 16);
        ub0.u[0] = *(const uint4*)(Brow + k);
        ub0.u[1] = *(const uint4*)(Brow + k + 16);
        ua1.u[0] = *(const uint4*)(Arow + k + 32);
        ua1.u[1] = *(const uint4*)(Arow + k + 48);
        ub1.u[0] = *(const uint4*)(Brow + k + 32);
        ub1.u[1] = *(const uint4*)(Brow + k + 48);
        acc0 = __builtin_amdgcn_wmma_f32_16x16x32_bf16(
            false, ua0.v, false, ub0.v, (short)0, acc0, false, false);
        acc1 = __builtin_amdgcn_wmma_f32_16x16x32_bf16(
            false, ua1.v, false, ub1.v, (short)0, acc1, false, false);
    }
    if (k < Kp) {                                // tail (Kp % 64 == 32)
        Frag ua, ub;
        ua.u[0] = *(const uint4*)(Arow + k);
        ua.u[1] = *(const uint4*)(Arow + k + 16);
        ub.u[0] = *(const uint4*)(Brow + k);
        ub.u[1] = *(const uint4*)(Brow + k + 16);
        acc0 = __builtin_amdgcn_wmma_f32_16x16x32_bf16(
            false, ua.v, false, ub.v, (short)0, acc0, false, false);
    }
#pragma unroll
    for (int i = 0; i < 8; ++i) acc0[i] += acc1[i];

    float* o = Out + ((size_t)b * C + (c0 + lm)) * Mp + m0 + 8 * half;
    union { v8f v; float4 f4[2]; } uo; uo.v = acc0;
    *(float4*)(o)     = uo.f4[0];
    *(float4*)(o + 4) = uo.f4[1];
}

// ---------------------------------------------------------------------------
// BatchNorm over (batch, Q) per channel + LeakyReLU, bf16 output.
// ---------------------------------------------------------------------------
__global__ void bn_reduce_kernel(const float* __restrict__ zT,  // [64][C][1024]
                                 float* __restrict__ stats, int C)
{
    const int c = blockIdx.x;
    float s = 0.0f, ss = 0.0f;
    for (int b = 0; b < 64; ++b) {
        const float* p = zT + ((size_t)b * C + c) * 1024;
        for (int q = threadIdx.x; q < 1024; q += 256) {
            const float v = p[q]; s += v; ss += v * v;
        }
    }
    __shared__ float sb[256], qb[256];
    sb[threadIdx.x] = s; qb[threadIdx.x] = ss;
    __syncthreads();
    for (int st = 128; st > 0; st >>= 1) {
        if (threadIdx.x < st) { sb[threadIdx.x] += sb[threadIdx.x + st];
                                qb[threadIdx.x] += qb[threadIdx.x + st]; }
        __syncthreads();
    }
    if (threadIdx.x == 0) {
        const float n = 65536.0f;
        const float m = sb[0] / n;
        const float var = qb[0] / n - m * m;
        stats[2 * c]     = m;
        stats[2 * c + 1] = rsqrtf(var + BN_EPS);
    }
}

__global__ void bn_apply_kernel(const float* __restrict__ zT,
                                const float* __restrict__ stats,
                                const float* __restrict__ g,
                                const float* __restrict__ be,
                                bf16* __restrict__ out, int C)
{
    const size_t i = (size_t)blockIdx.x * 256 + threadIdx.x;
    const size_t total = (size_t)64 * C * 1024;
    if (i >= total) return;
    const int c = (int)((i >> 10) % C);
    float v = (zT[i] - stats[2 * c]) * stats[2 * c + 1] * g[c] + be[c];
    v = (v > 0.0f) ? v : ALPHA * v;
    out[i] = (bf16)v;
}

// ---------------------------------------------------------------------------
// Channel mix W (Cin -> Cout) + bias, output transposed bf16, K-zero-padded.
// ---------------------------------------------------------------------------
template <int CIN, int COUT>
__global__ void mix_kernel(const float* __restrict__ wT,    // [64][CIN][Mdin]
                           int Mdin, int Tval,
                           const float* __restrict__ W,     // [CIN][COUT]
                           const float* __restrict__ bias,
                           bf16* __restrict__ out,          // [64][COUT][Tpout]
                           int Tpout)
{
    const int b = blockIdx.y;
    const int t = blockIdx.x * 128 + threadIdx.x;
    if (t >= Tpout) return;
    const bool valid = (t < Tval);
    float win[CIN];
#pragma unroll
    for (int ci = 0; ci < CIN; ++ci)
        win[ci] = valid ? wT[((size_t)b * CIN + ci) * Mdin + t] : 0.0f;
    for (int co = 0; co < COUT; ++co) {
        float a = valid ? bias[co] : 0.0f;
#pragma unroll
        for (int ci = 0; ci < CIN; ++ci) a += win[ci] * W[ci * COUT + co];
        out[((size_t)b * COUT + co) * Tpout + t] = (bf16)a;
    }
}

// ---------------------------------------------------------------------------
// Degree-wise norms over l-blocks -> h[64][256]  (h[b, d*64+c]).
// ---------------------------------------------------------------------------
__global__ void degree_norms_kernel(const float* __restrict__ wT,  // [64][64][96]
                                    float* __restrict__ h)         // [64][256]
{
    const int b = blockIdx.x, tid = threadIdx.x;
    const int d = tid >> 6, c = tid & 63;
    const int offs[4]  = {0, 1, 10, 35};
    const int sizes[4] = {1, 9, 25, 49};
    const float* p = wT + ((size_t)b * 64 + c) * 96;
    float s = 0.0f;
    for (int t = offs[d]; t < offs[d] + sizes[d]; ++t) { const float v = p[t]; s += v * v; }
    h[b * 256 + d * 64 + c] = sqrtf(s);
}

// ---------------------------------------------------------------------------
// FC + per-column BatchNorm (over batch of 64) + ReLU. One block per column.
// ---------------------------------------------------------------------------
__global__ void fc_bn_relu_kernel(const float* __restrict__ in,   // [64][K]
                                  const float* __restrict__ W,    // [K][N]
                                  const float* __restrict__ bias,
                                  const float* __restrict__ g,
                                  const float* __restrict__ be,
                                  float* __restrict__ out,        // [64][N]
                                  int K, int N)
{
    const int j = blockIdx.x;
    const int b = threadIdx.x;                 // 64 threads
    float v = bias[j];
    for (int k = 0; k < K; ++k) v += in[(size_t)b * K + k] * W[(size_t)k * N + j];
    __shared__ float buf[64];
    __shared__ float mv[2];
    buf[b] = v;
    __syncthreads();
    if (b == 0) {
        float s = 0.0f;
        for (int i = 0; i < 64; ++i) s += buf[i];
        const float m = s / 64.0f;
        float ss = 0.0f;
        for (int i = 0; i < 64; ++i) { const float d2 = buf[i] - m; ss += d2 * d2; }
        mv[0] = m; mv[1] = rsqrtf(ss / 64.0f + BN_EPS);
    }
    __syncthreads();
    float r = (v - mv[0]) * mv[1] * g[j] + be[j];
    out[(size_t)b * N + j] = (r > 0.0f) ? r : 0.0f;
}

__global__ void head_out_kernel(const float* __restrict__ h2,     // [64][256]
                                const float* __restrict__ Wout,   // [256][40]
                                const float* __restrict__ bout,
                                float* __restrict__ out)          // [64][40]
{
    const int b = blockIdx.x, j = threadIdx.x;   // 64 threads, j < 40 active
    __shared__ float lg[40];
    __shared__ float red[2];
    float v = 0.0f;
    if (j < 40) {
        v = bout[j];
        for (int k = 0; k < 256; ++k) v += h2[(size_t)b * 256 + k] * Wout[k * 40 + j];
        lg[j] = v;
    }
    __syncthreads();
    if (j == 0) {
        float mx = lg[0];
        for (int i = 1; i < 40; ++i) mx = fmaxf(mx, lg[i]);
        float s = 0.0f;
        for (int i = 0; i < 40; ++i) s += __expf(lg[i] - mx);
        red[0] = mx; red[1] = s;
    }
    __syncthreads();
    if (j < 40) out[b * 40 + j] = __expf(v - red[0]) / red[1];
}

// ---------------------------------------------------------------------------
static inline char* carve(char*& p, size_t bytes) {
    char* r = p;
    p += (bytes + 255) & ~(size_t)255;
    return r;
}

extern "C" void kernel_launch(void* const* d_in, const int* in_sizes, int n_in,
                              void* d_out, int out_size, void* d_ws, size_t ws_size,
                              hipStream_t stream) {
    (void)in_sizes; (void)n_in; (void)out_size; (void)ws_size;
    const float* x       = (const float*)d_in[0];
    const float* dirs    = (const float*)d_in[1];
    const float* A_sh    = (const float*)d_in[2];
    const float* D_eval0 = (const float*)d_in[3];
    const float* D_eval1 = (const float*)d_in[4];
    const float* D_eval2 = (const float*)d_in[5];
    const float* D_coef1 = (const float*)d_in[6];
    const float* D_coef2 = (const float*)d_in[7];
    const float* D_coefL = (const float*)d_in[8];
    const float* W0 = (const float*)d_in[9],  *b0 = (const float*)d_in[10];
    const float* W1 = (const float*)d_in[11], *b1 = (const float*)d_in[12];
    const float* W2 = (const float*)d_in[13], *b2 = (const float*)d_in[14];
    const float* g0 = (const float*)d_in[15], *be0 = (const float*)d_in[16];
    const float* g1 = (const float*)d_in[17], *be1 = (const float*)d_in[18];
    const float* g2 = (const float*)d_in[19], *be2 = (const float*)d_in[20];
    const float* Wfc1 = (const float*)d_in[21], *bfc1 = (const float*)d_in[22];
    const float* gfc1 = (const float*)d_in[23], *befc1 = (const float*)d_in[24];
    const float* Wfc2 = (const float*)d_in[25], *bfc2 = (const float*)d_in[26];
    const float* gfc2 = (const float*)d_in[27], *befc2 = (const float*)d_in[28];
    const float* Wout = (const float*)d_in[29], *bout = (const float*)d_in[30];
    float* out = (float*)d_out;

    // ---- workspace layout ----
    char* p = (char*)d_ws;
    bf16* A0bf = (bf16*)carve(p, (size_t)1024 * 288 * 2);
    bf16* A1bf = (bf16*)carve(p, (size_t)1024 * 192 * 2);
    bf16* A2bf = (bf16*)carve(p, (size_t)1024 * 96  * 2);
    bf16* C1bf = (bf16*)carve(p, (size_t)192 * 1024 * 2);   // 165 -> 192 (2-wave tiles)
    bf16* C2bf = (bf16*)carve(p, (size_t)96  * 1024 * 2);
    bf16* CLbf = (bf16*)carve(p, (size_t)96  * 1024 * 2);
    float* fbuf  = (float*)carve(p, (size_t)3 * 64 * 512 * 4);
    float* shbuf = (float*)carve(p, (size_t)64 * 36 * 3 * 4);
    bf16* yT1 = (bf16*)carve(p, (size_t)64 * 16 * 288 * 2);
    bf16* yT2 = (bf16*)carve(p, (size_t)64 * 32 * 192 * 2);
    bf16* yT3 = (bf16*)carve(p, (size_t)64 * 64 * 96  * 2);
    float* zT  = (float*)carve(p, (size_t)64 * 64 * 1024 * 4);
    bf16*  zbT = (bf16*) carve(p, (size_t)64 * 64 * 1024 * 2);
    float* wT  = (float*)carve(p, (size_t)64 * 64 * 192 * 4);
    float* stats = (float*)carve(p, 64 * 2 * 4);
    float* hbuf  = (float*)carve(p, (size_t)64 * 256 * 4);
    float* h1buf = (float*)carve(p, (size_t)64 * 512 * 4);
    float* h2buf = (float*)carve(p, (size_t)64 * 256 * 4);

    // ---- cast shared Wigner bases to padded bf16 ----
    auto cast = [&](const float* in, bf16* o, int M, int K, int Mp, int Kp) {
        int total = Mp * Kp;
        cast_pad_kernel<<<(total + 255) / 256, 256, 0, stream>>>(in, o, M, K, Mp, Kp);
    };
    cast(D_eval0, A0bf, 1024, 286, 1024, 288);
    cast(D_eval1, A1bf, 1024, 165, 1024, 192);
    cast(D_eval2, A2bf, 1024, 84,  1024, 96);
    cast(D_coef1, C1bf, 165, 1024, 192, 1024);
    cast(D_coef2, C2bf, 84,  1024, 96,  1024);
    cast(D_coefL, CLbf, 84,  1024, 96,  1024);

    // ---- shell embedding ----
    shell_density_kernel<<<dim3(32, 64, 3), 32, 0, stream>>>(x, dirs, fbuf);
    sh_project_kernel<<<dim3(64, 3), 64, 0, stream>>>(fbuf, A_sh, shbuf);
    mix0_kernel<<<64, 288, 0, stream>>>(shbuf, W0, b0, yT1);

    // ---- layer 0 ----
    gemm_bf16_kernel<<<dim3(32, 1, 64), 64, 0, stream>>>(A0bf, yT1, zT, 1024, 288, 16);
    bn_reduce_kernel<<<16, 256, 0, stream>>>(zT, stats, 16);
    bn_apply_kernel<<<(64 * 16 * 1024) / 256, 256, 0, stream>>>(zT, stats, g0, be0, zbT, 16);
    gemm_bf16_kernel<<<dim3(6, 1, 64), 64, 0, stream>>>(C1bf, zbT, wT, 192, 1024, 16);
    mix_kernel<16, 32><<<dim3(2, 64), 128, 0, stream>>>(wT, 192, 165, W1, b1, yT2, 192);

    // ---- layer 1 ----
    gemm_bf16_kernel<<<dim3(32, 2, 64), 64, 0, stream>>>(A1bf, yT2, zT, 1024, 192, 32);
    bn_reduce_kernel<<<32, 256, 0, stream>>>(zT, stats, 32);
    bn_apply_kernel<<<(64 * 32 * 1024) / 256, 256, 0, stream>>>(zT, stats, g1, be1, zbT, 32);
    gemm_bf16_kernel<<<dim3(3, 2, 64), 64, 0, stream>>>(C2bf, zbT, wT, 96, 1024, 32);
    mix_kernel<32, 64><<<dim3(1, 64), 128, 0, stream>>>(wT, 96, 84, W2, b2, yT3, 96);

    // ---- layer 2 ----
    gemm_bf16_kernel<<<dim3(32, 4, 64), 64, 0, stream>>>(A2bf, yT3, zT, 1024, 96, 64);
    bn_reduce_kernel<<<64, 256, 0, stream>>>(zT, stats, 64);
    bn_apply_kernel<<<(64 * 64 * 1024) / 256, 256, 0, stream>>>(zT, stats, g2, be2, zbT, 64);
    gemm_bf16_kernel<<<dim3(3, 4, 64), 64, 0, stream>>>(CLbf, zbT, wT, 96, 1024, 64);

    // ---- invariant pooling + MLP head ----
    degree_norms_kernel<<<64, 256, 0, stream>>>(wT, hbuf);
    fc_bn_relu_kernel<<<512, 64, 0, stream>>>(hbuf, Wfc1, bfc1, gfc1, befc1, h1buf, 256, 512);
    fc_bn_relu_kernel<<<256, 64, 0, stream>>>(h1buf, Wfc2, bfc2, gfc2, befc2, h2buf, 512, 256);
    head_out_kernel<<<64, 64, 0, stream>>>(h2buf, Wout, bout, out);
}
